// StructuredAttention_20306605375998
// MI455X (gfx1250) — compile-verified, matrix operand order checked by reading, not run on hardware
//
#include <hip/hip_runtime.h>
#include <hip/hip_bf16.h>
#include <math.h>

// ---------------------------------------------------------------------------
// StructuredAttention for MI455X (gfx1250, wave32, WMMA + TDM).
// Dense GEMMs: v_wmma_f32_16x16x32_bf16, f32 accumulation. Weight tiles are
// staged into LDS by the Tensor Data Mover (tensor_load_to_lds, TENSORcnt
// double-buffered); activations stream through VMEM b128 loads.
// The 16x (1024x1024) inversions run as L2-resident Gauss-Jordan
// (64 MB working set << 192 MB L2), one workgroup per matrix.
// ---------------------------------------------------------------------------

typedef __attribute__((ext_vector_type(16))) __bf16 v16bf;
typedef __attribute__((ext_vector_type(8)))  __bf16 v8bf;
typedef __attribute__((ext_vector_type(8)))  float  v8f;
typedef __attribute__((ext_vector_type(4)))  unsigned int u32x4;
typedef __attribute__((ext_vector_type(8)))  int i32x8;
typedef __attribute__((ext_vector_type(4)))  int i32x4;

#define BQ    16
#define NQ    1024
#define INTOT 2048
#define ATT   512
#define REPW  1536
#define FINW  3072

// LDS W tile: 64 rows x 32 bf16, padded to 40 elem (80 B) per row so that
// ds_load_b128 of 16 consecutive rows hits 16 distinct bank groups.
#define WROW 40

// inpd[c]: c<256 -> fwd[768+c] ; c in [256,512) -> bwd[768+(c-256)] = input[1536+c]
__device__ __forceinline__ int map_d(int c) { return (c < 256) ? (768 + c) : (1536 + c); }
// inpe[r]: r<768 -> fwd[r] ; r in [768,1536) -> bwd[r-768] = input[256+r]
__device__ __forceinline__ int map_e(int r) { return (r < 768) ? r : (256 + r); }

// ---------------- conversion / prep kernels ----------------

__global__ void k_convert_inpd(const float* __restrict__ in, __bf16* __restrict__ out) {
  long long idx = (long long)blockIdx.x * blockDim.x + threadIdx.x; // 16*1024*512
  int c = (int)(idx & 511);
  long long row = idx >> 9;
  out[idx] = (__bf16)in[row * INTOT + map_d(c)];
}

__global__ void k_f32_to_bf16(const float* __restrict__ in, __bf16* __restrict__ out, long long n) {
  long long idx = (long long)blockIdx.x * blockDim.x + threadIdx.x;
  if (idx < n) out[idx] = (__bf16)in[idx];
}

// WaT[e][d] = Wa[d][e]  (so the GEMM's row-major weight-fragment loads give A@Wa)
__global__ void k_transpose512(const float* __restrict__ in, __bf16* __restrict__ out) {
  int idx = blockIdx.x * blockDim.x + threadIdx.x; // 512*512
  int e = idx >> 9, d = idx & 511;
  out[idx] = (__bf16)in[d * 512 + e];
}

// ---------------- fr = exp(inpd @ wr) ----------------

__global__ __launch_bounds__(128) void k_fr(const float* __restrict__ in,
                                            const float* __restrict__ wr,
                                            float* __restrict__ fr) {
  __shared__ float red[128];
  long long row = blockIdx.x; // b*1024+n
  const float* src = in + row * INTOT;
  float s = 0.f;
  for (int c = threadIdx.x; c < ATT; c += 128) s += src[map_d(c)] * wr[c];
  red[threadIdx.x] = s;
  __syncthreads();
  for (int off = 64; off > 0; off >>= 1) {
    if ((int)threadIdx.x < off) red[threadIdx.x] += red[threadIdx.x + off];
    __syncthreads();
  }
  if (threadIdx.x == 0) fr[row] = expf(red[0]);
}

// ---------------- Tensor Data Mover: stage a 64x32 bf16 W tile into LDS ----------------
// D# packed per CDNA5 ISA 8.3/8.4: group0 {count=1, lds_addr, global_addr, type=2},
// group1 {data_size=2B, pad every 16 DWORDs (64B row) by 4 DWORDs (16B) -> 80B LDS
// row stride, tensor_dim0=32, tensor_dim1=64, tile 32x64, dim0_stride=ldw}.
// This toolchain's builtin takes 6 args (extra int32x8 group before cpol).
struct TdmG0 { unsigned long long q0, q1; };
struct TdmG1 { unsigned long long q0, q1, q2, q3; };

__device__ __forceinline__ void tdm_load_w_tile(unsigned lds_off, const __bf16* gptr, int ldw) {
  unsigned long long ga = (unsigned long long)(uintptr_t)gptr;
  TdmG0 g0s;
  g0s.q0 = 1ull | ((unsigned long long)lds_off << 32);         // count=1 | lds_addr
  g0s.q1 = (ga & 0x01FFFFFFFFFFFFFFull) | (2ull << 62);        // global_addr | type=2
  TdmG1 g1s;
  const unsigned long long flags =
      (1ull << 16) |   // data_size = 1 -> 2 bytes
      (1ull << 20) |   // pad_enable
      (3ull << 22) |   // pad_interval: 16 DWORDs (one 64B tile row)
      (3ull << 25);    // pad_amount:   4 DWORDs (16B)
  const unsigned long long td0 = 32, td1 = 64, tile0 = 32, tile1 = 64;
  const unsigned long long str0 = (unsigned long long)ldw;
  g1s.q0 = flags | ((td0 & 0xFFFFull) << 48);
  g1s.q1 = (td0 >> 16) | (td1 << 16) | (tile0 << 48);
  g1s.q2 = tile1 | ((str0 & 0xFFFFFFFFull) << 32);
  g1s.q3 = (str0 >> 32);                                       // dim1_stride unused (2D)
  u32x4 g0 = __builtin_bit_cast(u32x4, g0s);
  i32x8 g1 = __builtin_bit_cast(i32x8, g1s);
  i32x4 z4 = {0, 0, 0, 0};
  i32x8 z8 = {0, 0, 0, 0, 0, 0, 0, 0};
  __builtin_amdgcn_tensor_load_to_lds(g0, g1, z4, z4, z8, 0);
}

// ---------------- WMMA GEMM: Out = op(A @ W^T) ----------------
// Block = 256x64 output (4 waves x 64x64 wave tiles sharing one W tile).
// Fragment layout per CDNA5 ISA (16-bit A 16x32): lane<16 holds row r cols
// [k0..k0+7],[k0+16..k0+23]; lane>=16 holds [k0+8..k0+15],[k0+24..k0+31].
// mode 0: bf16 <- tanh(acc+bias)   mode 1: bf16 <- acc
// mode 2: f32  <- relu(acc+bias)   mode 3: f32  <- (gm==gn ? 0 : exp(acc))
__device__ __forceinline__ v16bf load_frag(const __bf16* base) {
  v8bf lo = *(const v8bf*)base;
  v8bf hi = *(const v8bf*)(base + 16);
  return __builtin_shufflevector(lo, hi, 0,1,2,3,4,5,6,7,8,9,10,11,12,13,14,15);
}

__global__ __launch_bounds__(128) void k_wmma_gemm(
    const __bf16* __restrict__ A, long long sA, int lda,
    const __bf16* __restrict__ W, long long sW, int ldw,
    const float* __restrict__ bias,
    void* __restrict__ Out, long long sO, int ldo,
    int Mb, int Nn, int K, int nB, int mode)
{
  (void)nB;
  __shared__ __bf16 lds_w[2][64 * WROW];
  const int lane = threadIdx.x & 31;
  const int wave = threadIdx.x >> 5;
  const int r = lane & 15;
  const int h = lane >> 4;

  const int tilesN = Nn >> 6;
  const int bpb = (Mb >> 8) * tilesN;              // blocks per batch
  const int b  = blockIdx.x / bpb;
  const int t  = blockIdx.x % bpb;
  const int m0 = (((t / tilesN) << 2) + wave) << 6;
  const int n0 = (t % tilesN) << 6;

  const __bf16* Ab = A + (long long)b * sA;
  const __bf16* Wb = W + (long long)b * sW + (long long)n0 * ldw;

  const unsigned lds_base = (unsigned)(uintptr_t)(void*)&lds_w[0][0];
  const unsigned lds_buf_bytes = 64 * WROW * 2;

  const v8f zero = {0.f,0.f,0.f,0.f,0.f,0.f,0.f,0.f};
  v8f acc[4][4];
#pragma unroll
  for (int p = 0; p < 4; ++p)
#pragma unroll
    for (int q = 0; q < 4; ++q) acc[p][q] = zero;

  // prologue: TDM fetch of the first W tile
  if (wave == 0) tdm_load_w_tile(lds_base, Wb, ldw);

  int cur = 0;
  for (int k0 = 0; k0 < K; k0 += 32) {
    const bool more = (k0 + 32) < K;
    if (wave == 0) {
      if (more) tdm_load_w_tile(lds_base + (cur ^ 1) * lds_buf_bytes, Wb + (k0 + 32), ldw);
      if (more) __builtin_amdgcn_s_wait_tensorcnt(1);   // current tile done, next in flight
      else      __builtin_amdgcn_s_wait_tensorcnt(0);
    }
    __syncthreads();                                    // W tile visible to all waves

    if (more)  // warm next A stripe (global_prefetch_b8)
      __builtin_prefetch(Ab + (long long)(m0 + r) * lda + (k0 + 32 + h * 8), 0, 1);

    v16bf af[4], wf[4];
#pragma unroll
    for (int p = 0; p < 4; ++p)
      af[p] = load_frag(Ab + (long long)(m0 + 16 * p + r) * lda + (k0 + h * 8));
#pragma unroll
    for (int q = 0; q < 4; ++q)
      wf[q] = load_frag(&lds_w[cur][(16 * q + r) * WROW + h * 8]);
#pragma unroll
    for (int p = 0; p < 4; ++p)
#pragma unroll
      for (int q = 0; q < 4; ++q)
        acc[p][q] = __builtin_amdgcn_wmma_f32_16x16x32_bf16(
            false, af[p], false, wf[q], (short)0, acc[p][q], false, false);

    __syncthreads();                                    // reads done before next TDM overwrite
    cur ^= 1;
  }

  // C/D layout: VGPR i, lane<16 -> (M=i, N=lane); lane>=16 -> (M=i+8, N=lane-16)
  if (mode <= 1) {
    __bf16* ob = (__bf16*)Out + (long long)b * sO;
#pragma unroll
    for (int p = 0; p < 4; ++p)
#pragma unroll
      for (int q = 0; q < 4; ++q) {
        const int gn = n0 + 16 * q + r;
        const float bv = (mode == 0) ? bias[gn] : 0.f;
#pragma unroll
        for (int i = 0; i < 8; ++i) {
          const int gm = m0 + 16 * p + 8 * h + i;
          float v = acc[p][q][i] + bv;
          if (mode == 0) v = tanhf(v);
          ob[(long long)gm * ldo + gn] = (__bf16)v;
        }
      }
  } else {
    float* ob = (float*)Out + (long long)b * sO;
#pragma unroll
    for (int p = 0; p < 4; ++p)
#pragma unroll
      for (int q = 0; q < 4; ++q) {
        const int gn = n0 + 16 * q + r;
#pragma unroll
        for (int i = 0; i < 8; ++i) {
          const int gm = m0 + 16 * p + 8 * h + i;
          float v = acc[p][q][i];
          if (mode == 2) { v += bias[gn]; v = v > 0.f ? v : 0.f; }
          else           { v = (gm == gn) ? 0.f : expf(v); }
          ob[(long long)gm * ldo + gn] = v;
        }
      }
  }
}

// ---------------- Laplacian build ----------------

__global__ void k_colsum(const float* __restrict__ Aij, float* __restrict__ Di) {
  int idx = blockIdx.x * blockDim.x + threadIdx.x; // 16384
  int b = idx >> 10, j = idx & 1023;
  const float* Ab = Aij + (long long)b * NQ * NQ;
  float s = 0.f;
  for (int i = 0; i < NQ; ++i) s += Ab[(long long)i * NQ + j];
  Di[idx] = s;
}

__global__ void k_build_LL(const float* __restrict__ Aij, const float* __restrict__ fr,
                           const float* __restrict__ Di, float* __restrict__ LL) {
  long long idx = (long long)blockIdx.x * blockDim.x + threadIdx.x; // 16*1024*1024
  int j = (int)(idx & 1023);
  long long t = idx >> 10;
  int i = (int)(t & 1023);
  int b = (int)(t >> 10);
  float v;
  if (i == 0)       v = fr[(b << 10) + j];
  else if (i == j)  v = Di[(b << 10) + i];   // Aij diag is zero
  else              v = -Aij[idx];
  LL[idx] = v;
}

// ---------------- Gauss-Jordan in-place inversion, one WG per matrix ----------------

__global__ __launch_bounds__(1024) void k_invert(float* __restrict__ LL) {
  float* A = LL + (long long)blockIdx.x * NQ * NQ;
  const int t  = threadIdx.x;
  const int tx = t & 31;   // column group (coalesced)
  const int ty = t >> 5;   // row group
  __shared__ float rowk[NQ];
  __shared__ float fcol[NQ];
  for (int k = 0; k < NQ; ++k) {
    float ip = 1.0f / A[(long long)k * NQ + k];
    float rv = A[(long long)k * NQ + t];
    rv = (t == k) ? ip : rv * ip;
    rowk[t] = rv;
    __syncthreads();
    A[(long long)k * NQ + t] = rv;           // scaled pivot row back to memory
    fcol[t] = A[(long long)t * NQ + k];      // pre-update factor column
    __syncthreads();
    for (int ii = 0; ii < 32; ++ii) {
      int i = ty + (ii << 5);
      if (i == k) continue;
      float f = fcol[i];
#pragma unroll 4
      for (int jj = 0; jj < 32; ++jj) {
        int j = tx + (jj << 5);
        float v = A[(long long)i * NQ + j];
        v = (j == k) ? (-f * rowk[k]) : (v - f * rowk[j]);
        A[(long long)i * NQ + j] = v;
      }
    }
    __syncthreads();
  }
}

// ---------------- post-inversion extraction ----------------

__global__ void k_post(const float* __restrict__ LL, const float* __restrict__ Aij,
                       const float* __restrict__ fr, float* __restrict__ d0,
                       float* __restrict__ dg, float* __restrict__ t0) {
  int idx = blockIdx.x * blockDim.x + threadIdx.x; // 16384
  int b = idx >> 10, j = idx & 1023;
  const float* Lb = LL + (long long)b * NQ * NQ;
  float dgv = Lb[(long long)j * NQ + j];           // diag of inverse
  d0[idx] = fr[idx] * Lb[(long long)j * NQ];       // fr * inv[:, 0]
  dg[idx] = dgv;
  t0[idx] = Aij[(long long)b * NQ * NQ + j] * dgv; // tmp[b,0,j]
}

// s0[b,r] = dg[b,0] * sum_{i>=1} Aij[b,i,0] * inpe[b,i,r]
__global__ __launch_bounds__(256) void k_s0(const float* __restrict__ Aij,
                                            const float* __restrict__ dg,
                                            const float* __restrict__ input,
                                            float* __restrict__ s0) {
  int b = blockIdx.x / 6;
  int r = (blockIdx.x % 6) * 256 + threadIdx.x;
  const float* Ab  = Aij  + (long long)b * NQ * NQ;
  const float* inb = input + (long long)b * NQ * INTOT;
  int me = map_e(r);
  float acc = 0.f;
  for (int i = 1; i < NQ; ++i)
    acc += Ab[(long long)i * NQ] * inb[(long long)i * INTOT + me];
  s0[b * REPW + r] = dg[b << 10] * acc;
}

// finp = [inpe | pinp] in bf16 for the final WMMA GEMM
__global__ void k_build_finp(const float* __restrict__ input, const float* __restrict__ ex,
                             const float* __restrict__ d0, const float* __restrict__ t0,
                             const float* __restrict__ s0, __bf16* __restrict__ finp) {
  long long idx = (long long)blockIdx.x * blockDim.x + threadIdx.x; // 16*1024*3072
  int r = (int)(idx % FINW);
  long long row = idx / FINW;               // b*1024 + j
  int b = (int)(row >> 10), j = (int)(row & 1023);
  float v;
  if (r < REPW) {
    v = input[row * INTOT + map_e(r)];
  } else {
    int rr = r - REPW;
    float e = ex[rr];
    if (j == 0) v = d0[b << 10] * e - s0[b * REPW + rr];
    else        v = d0[row] * e + t0[row] * input[(long long)(b << 10) * INTOT + map_e(rr)];
  }
  finp[idx] = (__bf16)v;
}

// ---------------------------------------------------------------------------

extern "C" void kernel_launch(void* const* d_in, const int* in_sizes, int n_in,
                              void* d_out, int out_size, void* d_ws, size_t ws_size,
                              hipStream_t stream) {
  const float* input = (const float*)d_in[0];
  const float* Wp    = (const float*)d_in[1];
  const float* bp    = (const float*)d_in[2];
  const float* Wc    = (const float*)d_in[3];
  const float* bc    = (const float*)d_in[4];
  const float* Wa    = (const float*)d_in[5];
  const float* wr    = (const float*)d_in[6];
  const float* Wz    = (const float*)d_in[7];
  const float* bz    = (const float*)d_in[8];
  const float* ex    = (const float*)d_in[9];

  char* ws = (char*)d_ws;
  // arena (bytes); finp aliases [inpd..Aij] which are dead by the time it is built
  const size_t o_inpd = 0;                 // 16 MiB  bf16 (16384 x 512)
  const size_t o_tp   = 16777216;          // 16 MiB
  const size_t o_tc   = 33554432;          // 16 MiB
  const size_t o_Ua   = 50331648;          // 16 MiB
  const size_t o_Aij  = 67108864;          // 64 MiB  f32 (16 x 1024 x 1024)
  const size_t o_finp = 0;                 // 96 MiB  bf16, aliases inpd/tp/tc/Ua/Aij
  const size_t o_LL   = 134217728;         // 64 MiB  f32
  const size_t o_Wz   = 201326592;         //  9 MiB  bf16
  const size_t o_Wp   = 210763776;
  const size_t o_Wc   = 211288064;
  const size_t o_WaT  = 211812352;
  const size_t o_fr   = 212336640;
  const size_t o_Di   = 212402176;
  const size_t o_dg   = 212467712;
  const size_t o_d0   = 212533248;
  const size_t o_t0   = 212598784;
  const size_t o_s0   = 212664320;

  __bf16* inpd_bf = (__bf16*)(ws + o_inpd);
  __bf16* tp_bf   = (__bf16*)(ws + o_tp);
  __bf16* tc_bf   = (__bf16*)(ws + o_tc);
  __bf16* Ua_bf   = (__bf16*)(ws + o_Ua);
  __bf16* finp_bf = (__bf16*)(ws + o_finp);
  __bf16* Wz_bf   = (__bf16*)(ws + o_Wz);
  __bf16* Wp_bf   = (__bf16*)(ws + o_Wp);
  __bf16* Wc_bf   = (__bf16*)(ws + o_Wc);
  __bf16* WaT_bf  = (__bf16*)(ws + o_WaT);
  float*  Aij     = (float*)(ws + o_Aij);
  float*  LL      = (float*)(ws + o_LL);
  float*  frv     = (float*)(ws + o_fr);
  float*  Di      = (float*)(ws + o_Di);
  float*  dg      = (float*)(ws + o_dg);
  float*  d0      = (float*)(ws + o_d0);
  float*  t0      = (float*)(ws + o_t0);
  float*  s0      = (float*)(ws + o_s0);

  // 1) precision conversion / packing
  k_convert_inpd<<<32768, 256, 0, stream>>>(input, inpd_bf);
  k_f32_to_bf16<<<1024, 256, 0, stream>>>(Wp, Wp_bf, 262144);
  k_f32_to_bf16<<<1024, 256, 0, stream>>>(Wc, Wc_bf, 262144);
  k_f32_to_bf16<<<18432, 256, 0, stream>>>(Wz, Wz_bf, 4718592);
  k_transpose512<<<1024, 256, 0, stream>>>(Wa, WaT_bf);

  // 2) tp = tanh(inpd @ Wp^T + bp), tc = tanh(inpd @ Wc^T + bc)   [WMMA bf16 + TDM]
  k_wmma_gemm<<<512, 128, 0, stream>>>(inpd_bf, 0, 512, Wp_bf, 0, 512, bp,
                                       (void*)tp_bf, 0, 512, 16384, 512, 512, 1, 0);
  k_wmma_gemm<<<512, 128, 0, stream>>>(inpd_bf, 0, 512, Wc_bf, 0, 512, bc,
                                       (void*)tc_bf, 0, 512, 16384, 512, 512, 1, 0);
  // 3) Ua = tp @ Wa  (via WaT)   [WMMA bf16 + TDM]
  k_wmma_gemm<<<512, 128, 0, stream>>>(tp_bf, 0, 512, WaT_bf, 0, 512, nullptr,
                                       (void*)Ua_bf, 0, 512, 16384, 512, 512, 1, 1);
  // 4) fr = exp(inpd @ wr)
  k_fr<<<16384, 128, 0, stream>>>(input, wr, frv);

  // 5) Aij = exp(Ua @ tc^T) with zero diagonal, batched over B   [WMMA bf16 + TDM]
  k_wmma_gemm<<<1024, 128, 0, stream>>>(Ua_bf, 1024LL*512, 512, tc_bf, 1024LL*512, 512,
                                        nullptr, (void*)Aij, 1024LL*1024, 1024,
                                        1024, 1024, 512, 16, 3);

  // 6) Laplacian, row-0 replacement, L2-resident Gauss-Jordan inversion
  k_colsum<<<64, 256, 0, stream>>>(Aij, Di);
  k_build_LL<<<65536, 256, 0, stream>>>(Aij, frv, Di, LL);
  k_invert<<<16, 1024, 0, stream>>>(LL);

  // 7) extract d0 / diag / row-0 and col-0 contractions (einsum collapses)
  k_post<<<64, 256, 0, stream>>>(LL, Aij, frv, d0, dg, t0);
  k_s0<<<96, 256, 0, stream>>>(Aij, dg, input, s0);

  // 8) assemble finp and final GEMM: out = relu(finp @ Wz^T + bz)   [WMMA bf16 + TDM]
  k_build_finp<<<196608, 256, 0, stream>>>(input, ex, d0, t0, s0, finp_bf);
  k_wmma_gemm<<<1536, 128, 0, stream>>>(finp_bf, 0, 3072, Wz_bf, 0, 3072, bz,
                                        d_out, 0, 1536, 16384, 1536, 3072, 1, 2);
}
